// SelfAttention_17781164606251
// MI455X (gfx1250) — compile-verified, matrix operand order checked by reading, not run on hardware
//
#include <hip/hip_runtime.h>
#include <hip/hip_bf16.h>
#include <math.h>

typedef __bf16 bf16;
typedef __attribute__((ext_vector_type(16))) __bf16 v16bf;
typedef __attribute__((ext_vector_type(8)))  __bf16 v8bf;
typedef __attribute__((ext_vector_type(8)))  float  v8f;

static constexpr int NB  = 4;     // batch
static constexpr int NC  = 512;   // channels
static constexpr int NQK = 64;    // q/k channels
static constexpr int NL  = 4096;  // sequence length

// Per-lane WMMA operand gather (identical index pattern for A and for B-as-Bt):
// lane holds row (lane&15); elements [kb..kb+7] and [kb+16..kb+23], kb = (lane<16?0:8).
__device__ __forceinline__ v16bf load_op(const bf16* row, int kb) {
  v8bf lo = *(const v8bf*)(row + kb);
  v8bf hi = *(const v8bf*)(row + kb + 16);
  return __builtin_shufflevector(lo, hi, 0,1,2,3,4,5,6,7,8,9,10,11,12,13,14,15);
}

__device__ __forceinline__ v8f wmma_bf16(v16bf a, v16bf b, v8f c) {
  return __builtin_amdgcn_wmma_f32_16x16x32_bf16(false, a, false, b, (short)0, c,
                                                 false, false);
}

// ---------------- fp32 -> bf16 weight conversion ----------------
__global__ void cvt_bf16_kernel(const float* __restrict__ src, bf16* __restrict__ dst, int n) {
  int i = blockIdx.x * blockDim.x + threadIdx.x;
  if (i < n) dst[i] = (bf16)src[i];
}

// ------------- x[B][C][L] f32 -> xT[B][L][C] bf16 (tiled transpose) -------------
__global__ void xT_kernel(const float* __restrict__ x, bf16* __restrict__ xT) {
  __shared__ float s[32][33];
  int b  = blockIdx.z;
  int j0 = blockIdx.x * 32, c0 = blockIdx.y * 32;
  int tx = threadIdx.x, ty = threadIdx.y; // 32 x 8
  const float* xb = x + (size_t)b * NC * NL;
  for (int r = ty; r < 32; r += 8)
    s[r][tx] = xb[(size_t)(c0 + r) * NL + j0 + tx];
  __syncthreads();
  bf16* xTb = xT + (size_t)b * NL * NC;
  for (int r = ty; r < 32; r += 8)
    xTb[(size_t)(j0 + r) * NC + c0 + tx] = (bf16)s[tx][r];
}

// ------------- qT / kT projection: qT[b][i][o] = sum_c xT[b][i][c]*W[o][c] + bias[o] -------------
__global__ void qk_proj_kernel(const bf16* __restrict__ xT,
                               const bf16* __restrict__ Wqb, const bf16* __restrict__ Wkb,
                               const float* __restrict__ bq, const float* __restrict__ bk,
                               bf16* __restrict__ qT, bf16* __restrict__ kT) {
  int lane = threadIdx.x & 31;
  int wid  = blockIdx.x * (blockDim.x >> 5) + (threadIdx.x >> 5);
  int ot = wid & 3;                 // o-tile (NQK/16 = 4)
  int it = (wid >> 2) & 255;        // i-tile (NL/16 = 256)
  int b  = (wid >> 10) & 3;
  int which = wid >> 12;            // 0=q 1=k
  const bf16* W    = which ? Wkb : Wqb;
  const float* bia = which ? bk  : bq;
  bf16* outT       = which ? kT  : qT;

  int kb = ((lane >> 4) & 1) * 8;
  int mn = lane & 15;
  const bf16* arow = xT + ((size_t)b * NL + it * 16 + mn) * NC;  // A: rows = i
  const bf16* brow = W  + (size_t)(ot * 16 + mn) * NC;           // B: rows = o (Bt layout)
  v8f acc = {};
  for (int kk = 0; kk < NC / 32; ++kk) {
    v16bf a  = load_op(arow + kk * 32, kb);
    v16bf bb = load_op(brow + kk * 32, kb);
    acc = wmma_bf16(a, bb, acc);
  }
  float bo = bia[ot * 16 + mn];                 // per-column bias
  bf16* orow = outT + ((size_t)b * NL + it * 16) * NQK + (ot * 16 + mn);
#pragma unroll
  for (int r = 0; r < 8; ++r)
    orow[(size_t)(kb + r) * NQK] = (bf16)(acc[r] + bo);
}

// ------------- v projection: v[b][co][j] = sum_c Wv[co][c]*xT[b][j][c] + bv[co] -------------
__global__ void v_proj_kernel(const bf16* __restrict__ xT, const bf16* __restrict__ Wvb,
                              const float* __restrict__ bv, bf16* __restrict__ vmat) {
  __shared__ __align__(16) bf16 st[4][16][16];
  int lane  = threadIdx.x & 31;
  int wslot = threadIdx.x >> 5;
  int wid   = blockIdx.x * 4 + wslot;
  int jt = wid & 255;               // j-tile
  int ct = (wid >> 8) & 31;         // c_out-tile
  int b  = wid >> 13;
  int kb = ((lane >> 4) & 1) * 8;
  int mn = lane & 15;
  const bf16* arow = Wvb + (size_t)(ct * 16 + mn) * NC;            // A: rows = c_out
  const bf16* brow = xT  + ((size_t)b * NL + jt * 16 + mn) * NC;   // B: rows = j
  v8f acc = {};
  for (int kk = 0; kk < NC / 32; ++kk) {
    v16bf a  = load_op(arow + kk * 32, kb);
    v16bf bb = load_op(brow + kk * 32, kb);
    acc = wmma_bf16(a, bb, acc);
  }
#pragma unroll
  for (int r = 0; r < 8; ++r)
    st[wslot][kb + r][mn] = (bf16)(acc[r] + bv[ct * 16 + kb + r]); // per-row bias
  __syncthreads();
  int row = lane >> 1, half = (lane & 1) * 8;                      // coalesced 16B store
  v8bf val = *(const v8bf*)&st[wslot][row][half];
  bf16* dst = vmat + ((size_t)b * NC + ct * 16 + row) * NL + jt * 16 + half;
  *(v8bf*)dst = val;
}

// ------------- fused flash attention + residual -------------
// block = 4 waves; block handles one 32-row i-block; wave w owns 128 channels.
// Each wave processes TWO 16-row i-tiles so every K/V operand load feeds 2x the WMMAs
// (~65 FLOP/byte from L2). Inner loop is phase-ordered to keep peak live VGPRs < 256.
__global__ void attn_kernel(const float* __restrict__ x,
                            const bf16* __restrict__ qT, const bf16* __restrict__ kT,
                            const bf16* __restrict__ vmat,
                            const float* __restrict__ gammap, float* __restrict__ out) {
  __shared__ float stf[4][16][17];
  int lane = threadIdx.x & 31;
  int w    = threadIdx.x >> 5;       // c-group 0..3
  int b    = blockIdx.y;
  int i0   = blockIdx.x * 32;        // 32 query rows per block
  int kb = ((lane >> 4) & 1) * 8;
  int mn = lane & 15;
  int cbase = w * 128;

  // Q operands (B of the S^T gemm), loop-invariant: lane holds row i of qT.
  v16bf q[2][2];
#pragma unroll
  for (int ti = 0; ti < 2; ++ti) {
    const bf16* qrow = qT + ((size_t)b * NL + i0 + ti * 16 + mn) * NQK;
    q[ti][0] = load_op(qrow, kb);        // o = 0..31
    q[ti][1] = load_op(qrow + 32, kb);   // o = 32..63
  }
  const bf16* kbas = kT   + (size_t)b * NL * NQK;
  const bf16* vb   = vmat + (size_t)b * NC * NL;

  float m_i[2]  = {-__builtin_inff(), -__builtin_inff()};
  float ssum[2] = {0.f, 0.f};
  v8f o[2][8] = {};

  for (int jb = 0; jb < NL; jb += 32) {
    // --- phase 1: scores. K operands shared by both i-tiles; ka* die after this. ---
    v8f s[2][2];
    {
      const bf16* k0 = kbas + (size_t)(jb + mn) * NQK;       // rows j=jb+0..15
      const bf16* k1 = kbas + (size_t)(jb + 16 + mn) * NQK;  // rows j=jb+16..31
      v16bf ka0 = load_op(k0, kb),      ka1 = load_op(k0 + 32, kb);
      v16bf ka2 = load_op(k1, kb),      ka3 = load_op(k1 + 32, kb);
      if (jb + 32 < NL) {  // prefetch next j-block's K rows (global_prefetch_b8)
        __builtin_prefetch(kbas + (size_t)(jb + 32 + mn) * NQK, 0, 0);
        __builtin_prefetch(kbas + (size_t)(jb + 48 + mn) * NQK, 0, 0);
      }
#pragma unroll
      for (int ti = 0; ti < 2; ++ti) {
        v8f t0 = {}, t1 = {};
        t0 = wmma_bf16(ka0, q[ti][0], t0);
        t0 = wmma_bf16(ka1, q[ti][1], t0);
        t1 = wmma_bf16(ka2, q[ti][0], t1);
        t1 = wmma_bf16(ka3, q[ti][1], t1);
        s[ti][0] = t0; s[ti][1] = t1;
      }
    }
    // lane holds row i = mn of tile ti: j = jb + kb + r (s[ti][0]), jb+16+kb+r (s[ti][1])

    // --- phase 2: online softmax; exp folded straight into bf16 P operand. ---
    v16bf P[2];
#pragma unroll
    for (int ti = 0; ti < 2; ++ti) {
      float bm = s[ti][0][0];
#pragma unroll
      for (int r = 0; r < 8; ++r) { bm = fmaxf(bm, s[ti][0][r]); bm = fmaxf(bm, s[ti][1][r]); }
      bm = fmaxf(bm, __shfl_xor(bm, 16, 32));        // combine lane pair (i, i+16)
      float m_old = m_i[ti];
      float m_new = fmaxf(m_old, bm);
      float lsum = 0.f;
#pragma unroll
      for (int r = 0; r < 8; ++r) {
        float p = __expf(s[ti][0][r] - m_new); lsum += p; P[ti][r] = (bf16)p;
      }
#pragma unroll
      for (int r = 0; r < 8; ++r) {
        float p = __expf(s[ti][1][r] - m_new); lsum += p; P[ti][8 + r] = (bf16)p;
      }
      lsum += __shfl_xor(lsum, 16, 32);
      float alpha = __expf(m_old - m_new);
      ssum[ti] = ssum[ti] * alpha + lsum;
      m_i[ti] = m_new;

      // rescale accumulators only when some row's max actually moved (wave-uniform branch)
      if (__any(m_new > m_old)) {
#pragma unroll
        for (int r = 0; r < 8; ++r) {
          float ar = __shfl(alpha, kb + r, 32);      // alpha of O-row (kb+r)
#pragma unroll
          for (int t = 0; t < 8; ++t) o[ti][t][r] *= ar;
        }
      }
    }

    // --- phase 3: O += P(16i x 32j) * V^T(32j x 16c); each V operand feeds both i-tiles ---
#pragma unroll
    for (int t = 0; t < 8; ++t) {
      const bf16* vrow = vb + (size_t)(cbase + t * 16 + mn) * NL + jb;
      v16bf bvv = load_op(vrow, kb);
      o[0][t] = wmma_bf16(P[0], bvv, o[0][t]);
      o[1][t] = wmma_bf16(P[1], bvv, o[1][t]);
    }
  }

  // --- normalize, residual, store (LDS-staged for contiguous 32B stores) ---
  float g = gammap[0];
  int crow = lane >> 1, ihalf = (lane & 1) * 8;
  for (int ti = 0; ti < 2; ++ti) {
    float inv[8];
#pragma unroll
    for (int r = 0; r < 8; ++r) inv[r] = 1.0f / __shfl(ssum[ti], kb + r, 32);
    for (int t = 0; t < 8; ++t) {
      __syncthreads();
#pragma unroll
      for (int r = 0; r < 8; ++r) stf[w][kb + r][mn] = o[ti][t][r] * inv[r];  // [i][c]
      __syncthreads();
      size_t base = ((size_t)b * NC + cbase + t * 16 + crow) * (size_t)NL
                    + i0 + ti * 16 + ihalf;
#pragma unroll
      for (int e = 0; e < 8; ++e)
        out[base + e] = x[base + e] + g * stf[w][ihalf + e][crow];
    }
  }
}

extern "C" void kernel_launch(void* const* d_in, const int* in_sizes, int n_in,
                              void* d_out, int out_size, void* d_ws, size_t ws_size,
                              hipStream_t stream) {
  const float* x     = (const float*)d_in[0];
  const float* Wq    = (const float*)d_in[1];
  const float* bq    = (const float*)d_in[2];
  const float* Wk    = (const float*)d_in[3];
  const float* bk    = (const float*)d_in[4];
  const float* Wv    = (const float*)d_in[5];
  const float* bv    = (const float*)d_in[6];
  const float* gamma = (const float*)d_in[7];
  float* out = (float*)d_out;

  char* p = (char*)d_ws;
  bf16* xT  = (bf16*)p; p += (size_t)NB * NL * NC * 2;    // 16 MB
  bf16* Wqb = (bf16*)p; p += (size_t)NQK * NC * 2;        // 64 KB
  bf16* Wkb = (bf16*)p; p += (size_t)NQK * NC * 2;        // 64 KB
  bf16* Wvb = (bf16*)p; p += (size_t)NC * NC * 2;         // 512 KB
  bf16* qTp = (bf16*)p; p += (size_t)NB * NL * NQK * 2;   // 2 MB
  bf16* kTp = (bf16*)p; p += (size_t)NB * NL * NQK * 2;   // 2 MB
  bf16* vp  = (bf16*)p; p += (size_t)NB * NC * NL * 2;    // 16 MB

  cvt_bf16_kernel<<<(NQK * NC + 255) / 256, 256, 0, stream>>>(Wq, Wqb, NQK * NC);
  cvt_bf16_kernel<<<(NQK * NC + 255) / 256, 256, 0, stream>>>(Wk, Wkb, NQK * NC);
  cvt_bf16_kernel<<<(NC * NC + 255) / 256, 256, 0, stream>>>(Wv, Wvb, NC * NC);
  xT_kernel<<<dim3(NL / 32, NC / 32, NB), dim3(32, 8), 0, stream>>>(x, xT);
  qk_proj_kernel<<<2048, 128, 0, stream>>>(xT, Wqb, Wkb, bq, bk, qTp, kTp);
  v_proj_kernel<<<8192, 128, 0, stream>>>(xT, Wvb, bv, vp);
  attn_kernel<<<dim3(NL / 32, NB), 128, 0, stream>>>(x, qTp, kTp, vp, gamma, out);
}
